// DisplaceNet2v2_3762391351822
// MI455X (gfx1250) — matrix operand order repaired and checked by reading, not compiled, not run on hardware
//
#include <hip/hip_runtime.h>

// ---------------------------------------------------------------------------
// DisplaceNet2v2 on gfx1250 (MI455X): bf16 WMMA GEMM pipeline, BN folded,
// weights staged to LDS via async global->LDS copies, B fragments register-
// resident across the cout loop. f32->bf16 via native hardware conversion.
// ---------------------------------------------------------------------------

#define KNN_K 10
#define NPTS  8192
#define NBATCH 4
#define COLS_PATCH (NBATCH * NPTS * KNN_K) // 327680 columns for patch net
#define COLS_DISP  (NBATCH * NPTS)         // 32768 columns for displace net
#define LDS_PAD 8                          // u16 pad per weight row (16B) -> bank rotation

typedef unsigned short u16;
typedef unsigned int   u32;
typedef unsigned long long u64;
typedef __attribute__((ext_vector_type(16))) __bf16 v16bf;
typedef __attribute__((ext_vector_type(8)))  float  v8f;

// Native hardware f32 -> bf16 conversion (RNE); backend selects v_cvt*bf16*.
__device__ __forceinline__ u16 f2bf(float f) {
  union { __bf16 h; u16 u; } cv;
  cv.h = (__bf16)f;
  return cv.u;
}
__device__ __forceinline__ float bf2f(u16 h) {
  return __uint_as_float(((u32)h) << 16);
}

// ---------------------------------------------------------------------------
// Weight folding: W'[o][c] = bf16(W[o][c] * s_o), s_o = gamma/sqrt(rv+eps);
// bias'[o] = (b_o - rm_o)*s_o + beta_o.  cin zero-padded to cinp.
// ---------------------------------------------------------------------------
__global__ __launch_bounds__(256) void fold_kernel(
    const float* __restrict__ W, const float* __restrict__ bsrc,
    const float* __restrict__ g, const float* __restrict__ be,
    const float* __restrict__ rm, const float* __restrict__ rv,
    int cout, int cin, int cinp, u16* __restrict__ Wdst, float* __restrict__ bdst)
{
  int i = blockIdx.x * 256 + threadIdx.x;
  if (i >= cout * cinp) return;
  int o = i / cinp, c = i - o * cinp;
  float s = 1.0f;
  float bb = bsrc[o];
  if (g != nullptr) {
    float inv = rsqrtf(rv[o] + 1e-5f);
    s = g[o] * inv;
    bb = (bsrc[o] - rm[o]) * s + be[o];
  }
  float w = (c < cin) ? W[o * cin + c] * s : 0.0f;
  Wdst[i] = f2bf(w);
  if (c == 0) bdst[o] = bb;
}

// ---------------------------------------------------------------------------
// Brute-force KNN (self included) + local coords -> act0 [col][32] bf16,
// channels 0..2 = (query - neighbor), 3..31 zero-padded for the K=32 chunk.
// ---------------------------------------------------------------------------
__global__ __launch_bounds__(256) void knn_kernel(
    const float* __restrict__ pc, u16* __restrict__ act0)
{
  __shared__ float sx[256], sy[256], sz[256];
  const int b = blockIdx.y;
  const int q = blockIdx.x * 256 + threadIdx.x;
  const float* P = pc + (size_t)b * NPTS * 3;
  const float qx = P[q * 3 + 0], qy = P[q * 3 + 1], qz = P[q * 3 + 2];

  float bd[KNN_K];
  int   bi[KNN_K];
#pragma unroll
  for (int k = 0; k < KNN_K; ++k) { bd[k] = 3.4e38f; bi[k] = 0; }

  for (int t = 0; t < NPTS; t += 256) {
    int j = t + threadIdx.x;
    sx[threadIdx.x] = P[j * 3 + 0];
    sy[threadIdx.x] = P[j * 3 + 1];
    sz[threadIdx.x] = P[j * 3 + 2];
    __syncthreads();
    for (int jj = 0; jj < 256; ++jj) {
      float dx = qx - sx[jj], dy = qy - sy[jj], dz = qz - sz[jj];
      float dd = dx * dx + dy * dy + dz * dz;
      if (dd < bd[KNN_K - 1]) {
        bd[KNN_K - 1] = dd; bi[KNN_K - 1] = t + jj;
#pragma unroll
        for (int k = KNN_K - 1; k > 0; --k) {
          if (bd[k] < bd[k - 1]) {
            float td = bd[k]; bd[k] = bd[k - 1]; bd[k - 1] = td;
            int   ti = bi[k]; bi[k] = bi[k - 1]; bi[k - 1] = ti;
          }
        }
      }
    }
    __syncthreads();
  }

  const size_t base = ((size_t)b * NPTS + q) * KNN_K;
#pragma unroll
  for (int k = 0; k < KNN_K; ++k) {
    int nb = bi[k];
    union { u16 h[32]; uint4 u[4]; } row;
#pragma unroll
    for (int c = 0; c < 32; ++c) row.h[c] = 0;
    row.h[0] = f2bf(qx - P[nb * 3 + 0]);
    row.h[1] = f2bf(qy - P[nb * 3 + 1]);
    row.h[2] = f2bf(qz - P[nb * 3 + 2]);
    uint4* dst = (uint4*)(act0 + (base + (size_t)k) * 32);
    dst[0] = row.u[0]; dst[1] = row.u[1]; dst[2] = row.u[2]; dst[3] = row.u[3];
  }
}

// ---------------------------------------------------------------------------
// bf16 WMMA GEMM, CIN templated:
//   Y[col][o] = act( sum_c W[o][c]*X[col][c] + bias[o] )
// - Folded weights async-copied global->LDS once per block (ASYNCcnt path),
//   LDS rows padded by 16B so A-fragment ds_load_b128 rotates across banks.
// - B fragments (activations) live in registers across the whole cout loop.
// X: [cols][xstride] bf16, W: [cout][CIN] bf16, Y: [cols][ystride] bf16.
// One wave -> 16 columns; 8 waves per block; grids are exact (no tail).
// ---------------------------------------------------------------------------
union Frag { uint4 u[2]; v16bf v; };

template<int CIN, bool LRELU>
__global__ __launch_bounds__(256) void gemm_lds_kernel(
    const u16* __restrict__ X, const u16* __restrict__ W,
    const float* __restrict__ bias, u16* __restrict__ Y,
    int cols, int cout, int xstride, int ystride)
{
  extern __shared__ u16 sW[];          // [cout][CIN + LDS_PAD]
  constexpr int NK  = CIN / 32;        // k-chunks of 32
  constexpr int CPR = CIN / 8;         // 16B chunks per weight row
  const int tid = threadIdx.x;

  // --- cooperative async global->LDS copy of the weight matrix ---
  {
    const u32 lds_base = (u32)(uintptr_t)(&sW[0]);
    const int nchunks = cout * CPR;    // 16B chunks total
    for (int c = tid; c < nchunks; c += 256) {
      int row = c / CPR, inrow = c - row * CPR;
      u32 ldst = lds_base + (u32)(row * (CIN + LDS_PAD) + inrow * 8) * 2u;
      u64 gsrc = (u64)(uintptr_t)(W + (size_t)row * CIN + inrow * 8);
      asm volatile("global_load_async_to_lds_b128 %0, %1, off"
                   :: "v"(ldst), "v"(gsrc) : "memory");
    }
    asm volatile("s_wait_asynccnt 0" ::: "memory");
  }
  __syncthreads();

  const int lane = tid & 31;
  const int wave = tid >> 5;
  const int half = lane >> 4;          // 0: lanes 0-15, 1: lanes 16-31
  const int l16  = lane & 15;
  const int col  = (blockIdx.x * 8 + wave) * 16 + l16;
  (void)cols;

  // --- B fragments: register-resident for the whole cout loop ---
  const u16* __restrict__ xcol = X + (size_t)col * xstride;
  Frag bfr[NK];
#pragma unroll
  for (int k = 0; k < NK; ++k) {
    const u16* bp = xcol + k * 32 + half * 8;
    bfr[k].u[0] = *(const uint4*)(bp);
    bfr[k].u[1] = *(const uint4*)(bp + 16);
  }

  for (int m = 0; m < cout; m += 16) {
    v8f acc = {};
    const u16* wrow = sW + (size_t)(m + l16) * (CIN + LDS_PAD);
#pragma unroll
    for (int k = 0; k < NK; ++k) {
      Frag a;   // A (16x32 bf16): lane<16 -> K 0..7 | 16..23, lane>=16 -> 8..15 | 24..31
      const u16* ap = wrow + k * 32 + half * 8;
      a.u[0] = *(const uint4*)(ap);        // ds_load_b128
      a.u[1] = *(const uint4*)(ap + 16);   // ds_load_b128
      acc = __builtin_amdgcn_wmma_f32_16x16x32_bf16(
          false, a.v, false, bfr[k].v, (short)0, acc, false, false);
    }
    // D (16x16 f32): this lane holds channels m+half*8 .. +7 of column `col`
    const int chBase = m + half * 8;
    u16 hs[8];
#pragma unroll
    for (int j = 0; j < 8; ++j) {
      float f = acc[j] + bias[chBase + j];
      if (LRELU) f = (f >= 0.0f) ? f : 0.2f * f;
      hs[j] = f2bf(f);
    }
    uint4 o4;
    o4.x = (u32)hs[0] | ((u32)hs[1] << 16);
    o4.y = (u32)hs[2] | ((u32)hs[3] << 16);
    o4.z = (u32)hs[4] | ((u32)hs[5] << 16);
    o4.w = (u32)hs[6] | ((u32)hs[7] << 16);
    *(uint4*)(Y + (size_t)col * ystride + chBase) = o4;
  }
}

// ---------------------------------------------------------------------------
// Positional encoding -> h [B*N][320] bf16, channels 0..2 pc, 3..32 sin,
// 33..62 cos, 319 zero pad. (63..318 written by reduce_kernel.)
// ---------------------------------------------------------------------------
__global__ __launch_bounds__(256) void posenc_kernel(
    const float* __restrict__ pc, u16* __restrict__ h)
{
  int i = blockIdx.x * 256 + threadIdx.x;   // over B*N
  if (i >= COLS_DISP) return;
  const float* p = pc + (size_t)i * 3;
  u16* row = h + (size_t)i * 320;
  float v[3] = { p[0], p[1], p[2] };
#pragma unroll
  for (int d = 0; d < 3; ++d) row[d] = f2bf(v[d]);
#pragma unroll
  for (int d = 0; d < 3; ++d) {
#pragma unroll
    for (int l = 0; l < 10; ++l) {
      float e = v[d] * (float)(1 << l);
      row[3  + d * 10 + l] = f2bf(__sinf(e));
      row[33 + d * 10 + l] = f2bf(__cosf(e));
    }
  }
  row[319] = 0;
}

// ---------------------------------------------------------------------------
// max/mean over K -> h channels 63..190 (max) and 191..318 (mean)
// ---------------------------------------------------------------------------
__global__ __launch_bounds__(256) void reduce_kernel(
    const u16* __restrict__ act4, u16* __restrict__ h)
{
  int i = blockIdx.x * 256 + threadIdx.x;   // over B*N*128
  if (i >= COLS_DISP * 128) return;
  int pt = i >> 7, ch = i & 127;
  const u16* a = act4 + (size_t)pt * KNN_K * 128 + ch;
  float mx = -3.4e38f, sm = 0.0f;
#pragma unroll
  for (int k = 0; k < KNN_K; ++k) {
    float f = bf2f(a[k * 128]);
    mx = fmaxf(mx, f);
    sm += f;
  }
  u16* row = h + (size_t)pt * 320;
  row[63  + ch] = f2bf(mx);
  row[191 + ch] = f2bf(sm * (1.0f / KNN_K));
}

// ---------------------------------------------------------------------------
// Final 3x64 projection (fp32) + residual: out = pc + 1e-3 * (V4 h3 + c4)
// ---------------------------------------------------------------------------
__global__ __launch_bounds__(256) void final_kernel(
    const float* __restrict__ pc, const u16* __restrict__ h3,
    const float* __restrict__ V4, const float* __restrict__ c4,
    float* __restrict__ out)
{
  int i = blockIdx.x * 256 + threadIdx.x;   // over B*N
  if (i >= COLS_DISP) return;
  const u16* hr = h3 + (size_t)i * 64;
  float hv[64];
#pragma unroll
  for (int c = 0; c < 64; ++c) hv[c] = bf2f(hr[c]);
#pragma unroll
  for (int o = 0; o < 3; ++o) {
    float s = c4[o];
    for (int c = 0; c < 64; ++c) s += V4[o * 64 + c] * hv[c];
    out[(size_t)i * 3 + o] = pc[(size_t)i * 3 + o] + 1e-3f * s;
  }
}

// ---------------------------------------------------------------------------
// Host-side orchestration
// ---------------------------------------------------------------------------
extern "C" void kernel_launch(void* const* d_in, const int* in_sizes, int n_in,
                              void* d_out, int out_size, void* d_ws, size_t ws_size,
                              hipStream_t stream) {
  (void)in_sizes; (void)n_in; (void)out_size; (void)ws_size;

  const float* pc = (const float*)d_in[0];
  // patch params
  const float* W1 = (const float*)d_in[1];  const float* b1 = (const float*)d_in[2];
  const float* g1 = (const float*)d_in[3];  const float* be1 = (const float*)d_in[4];
  const float* rm1 = (const float*)d_in[5]; const float* rv1 = (const float*)d_in[6];
  const float* W2 = (const float*)d_in[7];  const float* b2 = (const float*)d_in[8];
  const float* g2 = (const float*)d_in[9];  const float* be2 = (const float*)d_in[10];
  const float* rm2 = (const float*)d_in[11];const float* rv2 = (const float*)d_in[12];
  const float* W3 = (const float*)d_in[13]; const float* b3 = (const float*)d_in[14];
  const float* g3 = (const float*)d_in[15]; const float* be3 = (const float*)d_in[16];
  const float* rm3 = (const float*)d_in[17];const float* rv3 = (const float*)d_in[18];
  const float* W4 = (const float*)d_in[19]; const float* b4 = (const float*)d_in[20];
  // displace params
  const float* V1 = (const float*)d_in[21]; const float* c1 = (const float*)d_in[22];
  const float* h1g = (const float*)d_in[23];const float* h1b = (const float*)d_in[24];
  const float* h1m = (const float*)d_in[25];const float* h1v = (const float*)d_in[26];
  const float* V2 = (const float*)d_in[27]; const float* c2 = (const float*)d_in[28];
  const float* h2g = (const float*)d_in[29];const float* h2b = (const float*)d_in[30];
  const float* h2m = (const float*)d_in[31];const float* h2v = (const float*)d_in[32];
  const float* V3 = (const float*)d_in[33]; const float* c3 = (const float*)d_in[34];
  const float* h3g = (const float*)d_in[35];const float* h3b = (const float*)d_in[36];
  const float* h3m = (const float*)d_in[37];const float* h3v = (const float*)d_in[38];
  const float* V4 = (const float*)d_in[39]; const float* c4 = (const float*)d_in[40];

  float* out = (float*)d_out;

  // Workspace layout: two 84MiB ping-pong activation buffers + weight region
  char* ws = (char*)d_ws;
  u16* bufA = (u16*)ws;                                   // up to 327680*128 bf16
  u16* bufB = (u16*)(ws + 88080384ll);                    // ditto
  char* wreg = ws + 176160768ll;
  u16* W1f = (u16*)wreg;               // 64*32
  u16* W2f = W1f + 64 * 32;            // 128*64
  u16* W3f = W2f + 128 * 64;           // 128*128
  u16* W4f = W3f + 128 * 128;          // 128*128
  u16* V1f = W4f + 128 * 128;          // 256*320
  u16* V2f = V1f + 256 * 320;          // 128*256
  u16* V3f = V2f + 128 * 256;          // 64*128
  float* bpool = (float*)((char*)(V3f + 64 * 128));
  float* b1f = bpool;        // 64
  float* b2f = bpool + 64;   // 128
  float* b3f = bpool + 192;  // 128
  float* b4f = bpool + 320;  // 128
  float* d1f = bpool + 448;  // 256
  float* d2f = bpool + 704;  // 128
  float* d3f = bpool + 832;  // 64

  // ---- fold weights (BN into scale/bias, bf16, cin padded) ----
  auto foldN = [](int tot) { return (tot + 255) / 256; };
  fold_kernel<<<foldN(64 * 32),   256, 0, stream>>>(W1, b1, g1, be1, rm1, rv1, 64, 3, 32, W1f, b1f);
  fold_kernel<<<foldN(128 * 64),  256, 0, stream>>>(W2, b2, g2, be2, rm2, rv2, 128, 64, 64, W2f, b2f);
  fold_kernel<<<foldN(128 * 128), 256, 0, stream>>>(W3, b3, g3, be3, rm3, rv3, 128, 128, 128, W3f, b3f);
  fold_kernel<<<foldN(128 * 128), 256, 0, stream>>>(W4, b4, nullptr, nullptr, nullptr, nullptr, 128, 128, 128, W4f, b4f);
  fold_kernel<<<foldN(256 * 320), 256, 0, stream>>>(V1, c1, h1g, h1b, h1m, h1v, 256, 319, 320, V1f, d1f);
  fold_kernel<<<foldN(128 * 256), 256, 0, stream>>>(V2, c2, h2g, h2b, h2m, h2v, 128, 256, 256, V2f, d2f);
  fold_kernel<<<foldN(64 * 128),  256, 0, stream>>>(V3, c3, h3g, h3b, h3m, h3v, 64, 128, 128, V3f, d3f);

  // ---- KNN -> act0 [327680][32] bf16 in bufA ----
  knn_kernel<<<dim3(NPTS / 256, NBATCH), 256, 0, stream>>>(pc, bufA);

  // dynamic LDS bytes for a given (cout, cin)
  auto smemB = [](int cout, int cin) { return (size_t)cout * (cin + LDS_PAD) * 2; };

  // ---- patch feature net GEMMs ----
  gemm_lds_kernel<32, true><<<COLS_PATCH / 128, 256, smemB(64, 32), stream>>>(
      bufA, W1f, b1f, bufB, COLS_PATCH, 64, 32, 64);
  gemm_lds_kernel<64, true><<<COLS_PATCH / 128, 256, smemB(128, 64), stream>>>(
      bufB, W2f, b2f, bufA, COLS_PATCH, 128, 64, 128);
  gemm_lds_kernel<128, true><<<COLS_PATCH / 128, 256, smemB(128, 128), stream>>>(
      bufA, W3f, b3f, bufB, COLS_PATCH, 128, 128, 128);
  gemm_lds_kernel<128, false><<<COLS_PATCH / 128, 256, smemB(128, 128), stream>>>(
      bufB, W4f, b4f, bufA, COLS_PATCH, 128, 128, 128);

  // ---- build h [32768][320] in bufB: posenc + K-reduction of act4 ----
  posenc_kernel<<<COLS_DISP / 256, 256, 0, stream>>>(pc, bufB);
  reduce_kernel<<<(COLS_DISP * 128) / 256, 256, 0, stream>>>(bufA, bufB);

  // ---- displace net GEMMs ----
  gemm_lds_kernel<320, true><<<COLS_DISP / 128, 256, smemB(256, 320), stream>>>(
      bufB, V1f, d1f, bufA, COLS_DISP, 256, 320, 256);
  gemm_lds_kernel<256, true><<<COLS_DISP / 128, 256, smemB(128, 256), stream>>>(
      bufA, V2f, d2f, bufB, COLS_DISP, 128, 256, 128);
  gemm_lds_kernel<128, true><<<COLS_DISP / 128, 256, smemB(64, 128), stream>>>(
      bufB, V3f, d3f, bufA, COLS_DISP, 64, 128, 64);

  // ---- final 3x64 projection + residual ----
  final_kernel<<<COLS_DISP / 256, 256, 0, stream>>>(pc, bufA, V4, c4, out);
}
